// SimplifiedScaledDotProductAttention_3289944949035
// MI455X (gfx1250) — compile-verified, matrix operand order checked by reading, not run on hardware
//
#include <hip/hip_runtime.h>
#include <hip/hip_bf16.h>

typedef __attribute__((ext_vector_type(16))) _Float16 v16h;
typedef __attribute__((ext_vector_type(8)))  _Float16 v8h;
typedef __attribute__((ext_vector_type(8)))  float    v8f;

#define NQ 2048
#define NK 2048
#define DM 1024
#define NH 16
#define DK 64
#define SCALE 0.125f          /* 1/sqrt(64) */
#define L2E   1.44269504088896340736f

__device__ __forceinline__ v8f wmma_f16(v16h a, v16h b, v8f c) {
  return __builtin_amdgcn_wmma_f32_16x16x32_f16(false, a, false, b,
                                                (short)0, c, false, false);
}

// Load one 16-element f16 fragment (A or B operand, identical per-lane run
// pattern) from an fp32 row in global memory: elements 0..7 = row[lo..lo+7],
// 8..15 = row[lo+16..lo+23], scaled.
__device__ __forceinline__ v16h frag_from_f32row(const float* rowp, int lo, float s) {
  const float4 a0 = *(const float4*)(rowp + lo);
  const float4 a1 = *(const float4*)(rowp + lo + 4);
  const float4 b0 = *(const float4*)(rowp + lo + 16);
  const float4 b1 = *(const float4*)(rowp + lo + 20);
  v16h r;
  r[0]=(_Float16)(a0.x*s);  r[1]=(_Float16)(a0.y*s);
  r[2]=(_Float16)(a0.z*s);  r[3]=(_Float16)(a0.w*s);
  r[4]=(_Float16)(a1.x*s);  r[5]=(_Float16)(a1.y*s);
  r[6]=(_Float16)(a1.z*s);  r[7]=(_Float16)(a1.w*s);
  r[8]=(_Float16)(b0.x*s);  r[9]=(_Float16)(b0.y*s);
  r[10]=(_Float16)(b0.z*s); r[11]=(_Float16)(b0.w*s);
  r[12]=(_Float16)(b1.x*s); r[13]=(_Float16)(b1.y*s);
  r[14]=(_Float16)(b1.z*s); r[15]=(_Float16)(b1.w*s);
  return r;
}

// Same fragment pattern from an f16 LDS row (two ds_load_b128).
__device__ __forceinline__ v16h frag_from_ldsrow(const _Float16* rowp, int lo) {
  v8h a = *(const v8h*)(rowp + lo);
  v8h b = *(const v8h*)(rowp + lo + 16);
  return __builtin_shufflevector(a, b, 0,1,2,3,4,5,6,7,8,9,10,11,12,13,14,15);
}

#define KLDS_STRIDE 72   /* 144B: (stride/16B)=9 odd -> conflict-free b128 rows */
#define VLDS_STRIDE 40   /* 80B:  (stride/16B)=5 odd */

__global__ __launch_bounds__(256)
void attn_fwd(const float* __restrict__ Q, const float* __restrict__ K,
              const float* __restrict__ V, float* __restrict__ X) {
  __shared__ _Float16 Klds[32 * KLDS_STRIDE];   // [key][d]  f16
  __shared__ _Float16 VTlds[64 * VLDS_STRIDE];  // [d][key]  f16 (transposed)

  const int lane = threadIdx.x & 31;
  const int wave = threadIdx.x >> 5;
  const int qt = blockIdx.x & 15;          // 16 q-tiles of 128
  const int h  = (blockIdx.x >> 4) & 15;   // 16 heads
  const int b  = blockIdx.x >> 8;          // 2 batches
  const int lo = (lane & 16) ? 8 : 0;      // per-lane run offset within 32
  const int l15 = lane & 15;

  const int qrow = qt * 128 + wave * 16 + l15;
  const float* qptr = Q + (b * NQ + qrow) * DM + h * DK;

  // Qᵀ B-fragments (d-chunks 0-31 / 32-63), softmax scale folded in.
  const v16h qf0 = frag_from_f32row(qptr,      lo, SCALE);
  const v16h qf1 = frag_from_f32row(qptr + 32, lo, SCALE);

  v8f O[4];
  #pragma unroll
  for (int i = 0; i < 4; ++i) O[i] = (v8f){0,0,0,0,0,0,0,0};
  float m = -1.0e30f, l = 0.0f;

  for (int kb = 0; kb < NK; kb += 32) {
    __syncthreads();  // previous tile fully consumed
    // Cooperative fill: 32 keys x 64 dims; 256 threads x 2 float4.
    #pragma unroll
    for (int t = 0; t < 2; ++t) {
      const int idx = (threadIdx.x * 2 + t) * 4;   // linear f32 index < 2048
      const int row = idx >> 6;                    // key within tile
      const int col = idx & 63;                    // dim
      const float* g = K + (b * NK + kb + row) * DM + h * DK + col;
      const float4 kf = *(const float4*)g;
      _Float16* kd = &Klds[row * KLDS_STRIDE + col];
      kd[0]=(_Float16)kf.x; kd[1]=(_Float16)kf.y;
      kd[2]=(_Float16)kf.z; kd[3]=(_Float16)kf.w;
      const float4 vf = *(const float4*)(V + (b * NK + kb + row) * DM + h * DK + col);
      VTlds[(col+0) * VLDS_STRIDE + row] = (_Float16)vf.x;
      VTlds[(col+1) * VLDS_STRIDE + row] = (_Float16)vf.y;
      VTlds[(col+2) * VLDS_STRIDE + row] = (_Float16)vf.z;
      VTlds[(col+3) * VLDS_STRIDE + row] = (_Float16)vf.w;
    }
    __syncthreads();

    // Sᵀ = K · Qᵀ  (two 16key x 16q fragments, K-dim 64 = 2 WMMAs each)
    const _Float16* kr0 = &Klds[(l15)      * KLDS_STRIDE];
    const _Float16* kr1 = &Klds[(16 + l15) * KLDS_STRIDE];
    v16h k00 = frag_from_ldsrow(kr0,      lo);
    v16h k01 = frag_from_ldsrow(kr0 + 32, lo);
    v16h k10 = frag_from_ldsrow(kr1,      lo);
    v16h k11 = frag_from_ldsrow(kr1 + 32, lo);
    v8f s0 = (v8f){0,0,0,0,0,0,0,0};
    v8f s1 = (v8f){0,0,0,0,0,0,0,0};
    s0 = wmma_f16(k00, qf0, s0);  s0 = wmma_f16(k01, qf1, s0);
    s1 = wmma_f16(k10, qf0, s1);  s1 = wmma_f16(k11, qf1, s1);

    // Online softmax over this chunk's 32 keys (16 lane-local + 1 shuffle).
    float mc = -1.0e30f;
    #pragma unroll
    for (int r = 0; r < 8; ++r) { mc = fmaxf(mc, s0[r]); mc = fmaxf(mc, s1[r]); }
    mc = fmaxf(mc, __shfl_xor(mc, 16, 32));
    const float mnew = fmaxf(m, mc);
    const float corr = exp2f((m - mnew) * L2E);
    m = mnew;
    l *= corr;
    #pragma unroll
    for (int i = 0; i < 4; ++i) O[i] *= corr;

    // Pᵀ B-fragment feeds directly from the Sᵀ accumulator registers.
    v16h pf;
    float psum = 0.0f;
    #pragma unroll
    for (int r = 0; r < 8; ++r) {
      const float p0 = exp2f((s0[r] - m) * L2E);
      const float p1 = exp2f((s1[r] - m) * L2E);
      psum += p0 + p1;
      pf[r]     = (_Float16)p0;
      pf[8 + r] = (_Float16)p1;
    }
    l += psum;

    // Oᵀ += Vᵀ · Pᵀ  (4 d-groups of 16)
    #pragma unroll
    for (int gd = 0; gd < 4; ++gd) {
      v16h vfrag = frag_from_ldsrow(&VTlds[(gd * 16 + l15) * VLDS_STRIDE], lo);
      O[gd] = wmma_f16(vfrag, pf, O[gd]);
    }
  }

  // Finish: combine the two lane-half partial sums, normalize, store.
  l += __shfl_xor(l, 16, 32);
  const float inv = 1.0f / l;
  const int dhi = (lane & 16) ? 8 : 0;
  float* xrow = X + (b * NQ + qrow) * DM + h * DK;
  #pragma unroll
  for (int gd = 0; gd < 4; ++gd) {
    #pragma unroll
    for (int r = 0; r < 8; ++r)
      xrow[gd * 16 + r + dhi] = O[gd][r] * inv;
  }
}

// Y[m][n] = sum_k X[m][k] * W[n][k] + bias[n];  M=4096, N=1024, K=1024.
// Each wave: 32x32 tile (2x2 WMMA frags); block: 8 waves = 64(M) x 128(N).
__global__ __launch_bounds__(256)
void out_proj(const float* __restrict__ X, const float* __restrict__ W,
              const float* __restrict__ bias, float* __restrict__ Y) {
  const int lane = threadIdx.x & 31;
  const int wave = threadIdx.x >> 5;
  const int lo = (lane & 16) ? 8 : 0;
  const int l15 = lane & 15;
  const int bm = blockIdx.x & 63;
  const int bn = blockIdx.x >> 6;
  const int m0 = bm * 64 + (wave & 1) * 32;
  const int n0 = bn * 128 + (wave >> 1) * 32;

  const float* xr0 = X + (m0 + l15) * DM;
  const float* xr1 = X + (m0 + 16 + l15) * DM;
  const float* wr0 = W + (n0 + l15) * DM;
  const float* wr1 = W + (n0 + 16 + l15) * DM;

  v8f C[2][2];
  #pragma unroll
  for (int i = 0; i < 2; ++i)
    #pragma unroll
    for (int j = 0; j < 2; ++j) C[i][j] = (v8f){0,0,0,0,0,0,0,0};

  for (int kc = 0; kc < DM; kc += 32) {
    v16h a0 = frag_from_f32row(xr0 + kc, lo, 1.0f);
    v16h a1 = frag_from_f32row(xr1 + kc, lo, 1.0f);
    v16h b0 = frag_from_f32row(wr0 + kc, lo, 1.0f);
    v16h b1 = frag_from_f32row(wr1 + kc, lo, 1.0f);
    C[0][0] = wmma_f16(a0, b0, C[0][0]);
    C[0][1] = wmma_f16(a0, b1, C[0][1]);
    C[1][0] = wmma_f16(a1, b0, C[1][0]);
    C[1][1] = wmma_f16(a1, b1, C[1][1]);
  }

  const float bias0 = bias[n0 + l15];
  const float bias1 = bias[n0 + 16 + l15];
  const int mhi = (lane & 16) ? 8 : 0;
  #pragma unroll
  for (int mi = 0; mi < 2; ++mi) {
    #pragma unroll
    for (int ni = 0; ni < 2; ++ni) {
      const float bb = ni ? bias1 : bias0;
      #pragma unroll
      for (int r = 0; r < 8; ++r) {
        const int row = m0 + mi * 16 + r + mhi;
        const int col = n0 + ni * 16 + l15;
        Y[row * DM + col] = C[mi][ni][r] + bb;
      }
    }
  }
}

extern "C" void kernel_launch(void* const* d_in, const int* in_sizes, int n_in,
                              void* d_out, int out_size, void* d_ws, size_t ws_size,
                              hipStream_t stream) {
  (void)in_sizes; (void)n_in; (void)out_size; (void)ws_size;
  const float* Q  = (const float*)d_in[0];
  const float* K  = (const float*)d_in[1];
  const float* V  = (const float*)d_in[2];
  const float* Wo = (const float*)d_in[3];
  const float* bo = (const float*)d_in[4];
  float* X = (float*)d_ws;            // (b, q, h*d_k) fp32 intermediate
  float* Y = (float*)d_out;

  // 2 batches * 16 heads * 16 q-tiles = 512 blocks
  attn_fwd<<<dim3(512), dim3(256), 0, stream>>>(Q, K, V, X);
  // (4096/64) * (1024/128) = 64 * 8 = 512 blocks
  out_proj<<<dim3(512), dim3(256), 0, stream>>>(X, Wo, bo, Y);
}
